// CustomTransformerDecoder_38963943309550
// MI455X (gfx1250) — matrix operand order repaired and checked by reading, not compiled, not run on hardware
//
#include <hip/hip_runtime.h>
#include <hip/hip_bf16.h>
#include <math.h>

// ---------------- problem constants ----------------
#define BB 8
#define SS 1024
#define EE 512
#define HH 8
#define DD 64
#define LL 4
#define NHH 2048
#define VV 32000
#define SCALE 0.03125f   // S^-0.5 = 1024^-0.5 = 1/32 (reference scales by seq_len^-0.5)

// ---------------- WMMA helpers ----------------
typedef __attribute__((ext_vector_type(16))) __bf16 v16bf;
typedef __attribute__((ext_vector_type(8)))  float  v8f;
typedef unsigned int u32x4 __attribute__((ext_vector_type(4)));
typedef int          i32x8 __attribute__((ext_vector_type(8)));
typedef int          i32x4 __attribute__((ext_vector_type(4)));

union Frag { v16bf v; uint4 u4[2]; };   // 16 bf16 = 32 bytes = 2 x b128

__device__ inline v8f wmma_bf16(const Frag& a, const Frag& b, v8f c) {
  return __builtin_amdgcn_wmma_f32_16x16x32_bf16(
      /*neg_a=*/false, a.v, /*neg_b=*/false, b.v,
      /*c_mod=*/(short)0, c, /*reuse_a=*/false, /*reuse_b=*/false);
}

// CDNA5 async global->LDS copy (ASYNCcnt-tracked), 16B per lane.
// LDS offset = low 32 bits of the generic pointer (LDS aperture keeps the
// byte offset in addr[31:0] per the CDNA5 flat-addressing rules).
__device__ inline void async_copy_b128(const float* gp, const float* lds_ptr) {
  unsigned lds = (unsigned)(unsigned long long)lds_ptr;
  asm volatile("global_load_async_to_lds_b128 %0, %1, off"
               :: "v"(lds), "v"(gp) : "memory");
}
__device__ inline void async_wait0() {
  asm volatile("s_wait_asynccnt 0" ::: "memory");
}

// ---- Tensor Data Mover: 2D f32 tile (tile0 x tile1, row stride in elems) ----
// D# packing per CDNA5 ISA:
//  g0: [1:0]=count=1 | lds_addr[63:32] | global_addr[120:64] | type=2 @127:126
//  g1: data_size=2(4B)@[17:16] | tensor_dim0@[79:48] | tensor_dim1@[111:80] |
//      tile_dim0@[127:112] | tile_dim1@[143:128] | tensor_dim0_stride@[207:160]
// This toolchain exposes the 6-arg builtin:
//   (u32x4 g0, i32x8 g1, i32x4 g2, i32x4 g3, i32x8 extra, i32 cpol)
#if __has_builtin(__builtin_amdgcn_tensor_load_to_lds)
#define HAVE_TDM 1
__device__ inline void tdm_issue_2d(const float* gaddr, const float* lds_ptr,
                                    unsigned tdim0, unsigned tdim1,
                                    unsigned tile0, unsigned tile1,
                                    unsigned stride_elems) {
  unsigned long long ga = (unsigned long long)gaddr;
  unsigned lds = (unsigned)(unsigned long long)lds_ptr;
  u32x4 g0;
  g0[0] = 1u;                                               // count=1 (user D#)
  g0[1] = lds;                                              // lds_addr (bytes)
  g0[2] = (unsigned)(ga & 0xFFFFFFFFu);                     // global_addr lo
  g0[3] = (unsigned)((ga >> 32) & 0x01FFFFFFu) | (2u << 30);// ga hi | type=2
  i32x8 g1;
  g1[0] = (int)(2u << 16);                                  // data_size=4B
  g1[1] = (int)((tdim0 & 0xFFFFu) << 16);                   // tdim0[15:0]
  g1[2] = (int)((tdim0 >> 16) | ((tdim1 & 0xFFFFu) << 16)); // tdim0 hi|tdim1 lo
  g1[3] = (int)((tdim1 >> 16) | (tile0 << 16));             // tdim1 hi|tile0
  g1[4] = (int)(tile1 & 0xFFFFu);                           // tile1; tile2=0
  g1[5] = (int)stride_elems;                                // dim0 stride lo
  g1[6] = 0;
  g1[7] = 0;
  i32x4 z4 = (i32x4){0, 0, 0, 0};
  i32x8 z8 = (i32x8){0, 0, 0, 0, 0, 0, 0, 0};
  __builtin_amdgcn_tensor_load_to_lds(g0, g1, z4, z4, z8, 0);
}
#else
#define HAVE_TDM 0
#endif

// ---------------- embedding + positional encoding ----------------
__global__ __launch_bounds__(256) void embed_kernel(
    const int* __restrict__ x, const float* __restrict__ emb,
    float* __restrict__ h) {
  long n = (long)blockIdx.x * 256 + threadIdx.x;     // over B*S*E
  int e  = (int)(n % EE);
  long bs = n / EE;
  int s  = (int)(bs % SS);
  int tok = x[bs];
  float f = __expf(-(float)(e & ~1) * (9.210340371976184f / (float)EE));
  float pe = (e & 1) ? __cosf((float)s * f) : __sinf((float)s * f);
  h[n] = emb[(long)tok * EE + e] + pe;
}

// ---------------- layernorm (one row of E=512 per 256-thread block) ---------
__global__ __launch_bounds__(256) void ln_kernel(
    const float* __restrict__ x, const float* __restrict__ g,
    const float* __restrict__ bta, float* __restrict__ o) {
  __shared__ float red[256];
  long row = blockIdx.x;
  const float* xr = x + row * EE;
  int t = threadIdx.x;
  float a0 = xr[t], a1 = xr[t + 256];
  red[t] = a0 + a1;
  __syncthreads();
  for (int st = 128; st > 0; st >>= 1) {
    if (t < st) red[t] += red[t + st];
    __syncthreads();
  }
  float mu = red[0] * (1.0f / EE);
  __syncthreads();
  float d0 = a0 - mu, d1 = a1 - mu;
  red[t] = d0 * d0 + d1 * d1;
  __syncthreads();
  for (int st = 128; st > 0; st >>= 1) {
    if (t < st) red[t] += red[t + st];
    __syncthreads();
  }
  float r = rsqrtf(red[0] * (1.0f / EE) + 1e-5f);
  o[row * EE + t]       = d0 * r * g[t]       + bta[t];
  o[row * EE + t + 256] = d1 * r * g[t + 256] + bta[t + 256];
}

// ---------------- generic bf16 WMMA GEMM ----------------
// C[M,N] = A[M,K] * B(K,N) (+bias) (+relu).
//  - A tile (64x32 f32) staged by one TENSOR_LOAD_TO_LDS per K-step (TDM
//    zero-fills rows beyond M via tensor_dim1 = M-m0); async-copy fallback.
//  - B tile staged transposed as bf16 so fragment loads are 2x ds_load_b128.
// BMODE 0: B row-major [K,N].  BMODE 1: per-layer W[q|k|v] block [H][E][D]
// viewed as [K=E, N=H*D].
template <int BMODE, bool BIAS, bool RELU>
__global__ __launch_bounds__(128) void gemm_kernel(
    const float* __restrict__ A, const float* __restrict__ Bm,
    const float* __restrict__ bias, float* __restrict__ C,
    int M, int N, int K) {
  __shared__ float  Asf[64][32];   // f32 A tile (TDM/async-filled)
  __shared__ __bf16 Bst[64][32];   // B tile TRANSPOSED: [n][k]
  const int tid  = threadIdx.x;
  const int lane = tid & 31;
  const int wave = tid >> 5;
  const int m0 = blockIdx.y * 64;
  const int n0 = blockIdx.x * 64;
  const int wm = (wave & 1) * 32;
  const int wn = (wave >> 1) * 32;

  v8f acc[2][2];
  for (int i = 0; i < 2; ++i)
    for (int j = 0; j < 2; ++j) acc[i][j] = (v8f){};

  for (int k0 = 0; k0 < K; k0 += 32) {
    // ---- stage A tile (64 rows x 32 f32) via the Tensor Data Mover
#if HAVE_TDM
    if (tid == 0)
      tdm_issue_2d(A + (long)m0 * K + k0, &Asf[0][0],
                   (unsigned)(K - k0), (unsigned)(M - m0),
                   32u, 64u, (unsigned)K);
#else
    for (int tt = tid; tt < 512; tt += 128) {
      int r = tt >> 3, c4 = (tt & 7) << 2;
      int m = m0 + r;
      if (m < M)
        async_copy_b128(A + (long)m * K + k0 + c4, &Asf[r][c4]);
    }
#endif
    // ---- stage B tile transposed (vector f32 reads, bf16 stores);
    //      overlaps with the TDM transfer
    for (int tt = tid; tt < 512; tt += 128) {
      int r = tt >> 4, c4 = (tt & 15) << 2;        // k row, n column group
      int k = k0 + r, n = n0 + c4;
      float4 f = {0.f, 0.f, 0.f, 0.f};
      if (n < N) {
        if (BMODE == 0)
          f = *(const float4*)(Bm + (long)k * N + n);
        else
          f = *(const float4*)(Bm + (long)(n >> 6) * K * 64 + (long)k * 64 + (n & 63));
      }
      Bst[c4 + 0][r] = (__bf16)f.x;
      Bst[c4 + 1][r] = (__bf16)f.y;
      Bst[c4 + 2][r] = (__bf16)f.z;
      Bst[c4 + 3][r] = (__bf16)f.w;
    }
#if HAVE_TDM
    if (tid == 0) __builtin_amdgcn_s_wait_tensorcnt(0);
#else
    async_wait0();
#endif
    __syncthreads();

    if (k0 + 32 < K)   // speculative prefetch of next A tile
      __builtin_prefetch(A + (long)(m0 + (tid >> 1)) * K + k0 + 32, 0, 0);

    // ---- fragments
    Frag af[2], bfr[2];
    for (int rt = 0; rt < 2; ++rt) {
      int row = wm + rt * 16 + (lane & 15);
      int kb  = (lane >> 4) * 8;
      const float* ar = &Asf[row][0];
      float4 f0 = *(const float4*)(ar + kb);
      float4 f1 = *(const float4*)(ar + kb + 4);
      float4 g0 = *(const float4*)(ar + 16 + kb);
      float4 g1 = *(const float4*)(ar + 16 + kb + 4);
      af[rt].v[0]  = (__bf16)f0.x; af[rt].v[1]  = (__bf16)f0.y;
      af[rt].v[2]  = (__bf16)f0.z; af[rt].v[3]  = (__bf16)f0.w;
      af[rt].v[4]  = (__bf16)f1.x; af[rt].v[5]  = (__bf16)f1.y;
      af[rt].v[6]  = (__bf16)f1.z; af[rt].v[7]  = (__bf16)f1.w;
      af[rt].v[8]  = (__bf16)g0.x; af[rt].v[9]  = (__bf16)g0.y;
      af[rt].v[10] = (__bf16)g0.z; af[rt].v[11] = (__bf16)g0.w;
      af[rt].v[12] = (__bf16)g1.x; af[rt].v[13] = (__bf16)g1.y;
      af[rt].v[14] = (__bf16)g1.z; af[rt].v[15] = (__bf16)g1.w;
    }
    for (int nt = 0; nt < 2; ++nt) {
      int col  = wn + nt * 16 + (lane & 15);
      int kb8 = (lane >> 4) * 8;
      bfr[nt].u4[0] = *(const uint4*)&Bst[col][kb8];
      bfr[nt].u4[1] = *(const uint4*)&Bst[col][16 + kb8];
    }
    for (int rt = 0; rt < 2; ++rt)
      for (int nt = 0; nt < 2; ++nt)
        acc[rt][nt] = wmma_bf16(af[rt], bfr[nt], acc[rt][nt]);
    __syncthreads();
  }

  // ---- store C (f32 16x16 layout)
  for (int rt = 0; rt < 2; ++rt)
    for (int nt = 0; nt < 2; ++nt) {
      int colb = n0 + wn + nt * 16 + (lane & 15);
      for (int i = 0; i < 8; ++i) {
        int row = m0 + wm + rt * 16 + ((lane >> 4) ? i + 8 : i);
        if (row < M && colb < N) {
          float v = acc[rt][nt][i];
          if (BIAS) v += bias[colb];
          if (RELU) v = fmaxf(v, 0.0f);
          C[(long)row * N + colb] = v;
        }
      }
    }
}

// ---------------- fused causal attention ----------------
__global__ __launch_bounds__(128) void attn_kernel(
    const float* __restrict__ q, const float* __restrict__ k,
    const float* __restrict__ v, float* __restrict__ hres,
    float* __restrict__ map_out) {
  extern __shared__ float smem[];
  float* sc    = smem;                 // [32][1024]
  float* sm_m  = smem + 32 * 1024;     // [32]
  float* sm_il = sm_m + 32;            // [32]
  float* Vt    = sm_il + 32;           // [32][64] async-staged V tile

  const int b  = blockIdx.z;
  const int hh = blockIdx.y;
  const int q0 = blockIdx.x * 32;
  const int keyLen = q0 + 32;          // keys beyond are fully masked: skipped
  const int tid  = threadIdx.x;
  const int lane = tid & 31;
  const int wave = tid >> 5;

  // ---- preload q fragments (vector global loads, native bf16 cvt)
  Frag aq[2][2];
  for (int rt = 0; rt < 2; ++rt) {
    int qidx = q0 + rt * 16 + (lane & 15);
    int kb = (lane >> 4) * 8;
    const float* qp = q + ((long)(b * SS + qidx)) * EE + hh * DD;
    for (int c = 0; c < 2; ++c) {
      float4 f0 = *(const float4*)(qp + c * 32 + kb);
      float4 f1 = *(const float4*)(qp + c * 32 + kb + 4);
      float4 g0 = *(const float4*)(qp + c * 32 + 16 + kb);
      float4 g1 = *(const float4*)(qp + c * 32 + 16 + kb + 4);
      Frag& a = aq[rt][c];
      a.v[0]  = (__bf16)f0.x; a.v[1]  = (__bf16)f0.y;
      a.v[2]  = (__bf16)f0.z; a.v[3]  = (__bf16)f0.w;
      a.v[4]  = (__bf16)f1.x; a.v[5]  = (__bf16)f1.y;
      a.v[6]  = (__bf16)f1.z; a.v[7]  = (__bf16)f1.w;
      a.v[8]  = (__bf16)g0.x; a.v[9]  = (__bf16)g0.y;
      a.v[10] = (__bf16)g0.z; a.v[11] = (__bf16)g0.w;
      a.v[12] = (__bf16)g1.x; a.v[13] = (__bf16)g1.y;
      a.v[14] = (__bf16)g1.z; a.v[15] = (__bf16)g1.w;
    }
  }

  // ---- scores: key tiles of 16, round-robin over waves
  int nkt = keyLen >> 4;
  for (int kt = wave; kt < nkt; kt += 4) {
    Frag bk[2];
    int key  = kt * 16 + (lane & 15);
    int kb8 = (lane >> 4) * 8;
    const float* kp = k + ((long)(b * SS + key)) * EE + hh * DD;
    for (int c = 0; c < 2; ++c) {
      float4 f0 = *(const float4*)(kp + c * 32 + kb8);
      float4 f1 = *(const float4*)(kp + c * 32 + kb8 + 4);
      float4 f2 = *(const float4*)(kp + c * 32 + 16 + kb8);
      float4 f3 = *(const float4*)(kp + c * 32 + 16 + kb8 + 4);
      Frag& bb = bk[c];
      bb.v[0]  = (__bf16)f0.x; bb.v[1]  = (__bf16)f0.y;
      bb.v[2]  = (__bf16)f0.z; bb.v[3]  = (__bf16)f0.w;
      bb.v[4]  = (__bf16)f1.x; bb.v[5]  = (__bf16)f1.y;
      bb.v[6]  = (__bf16)f1.z; bb.v[7]  = (__bf16)f1.w;
      bb.v[8]  = (__bf16)f2.x; bb.v[9]  = (__bf16)f2.y;
      bb.v[10] = (__bf16)f2.z; bb.v[11] = (__bf16)f2.w;
      bb.v[12] = (__bf16)f3.x; bb.v[13] = (__bf16)f3.y;
      bb.v[14] = (__bf16)f3.z; bb.v[15] = (__bf16)f3.w;
    }
    for (int rt = 0; rt < 2; ++rt) {
      v8f s = (v8f){};
      s = wmma_bf16(aq[rt][0], bk[0], s);
      s = wmma_bf16(aq[rt][1], bk[1], s);
      int kcol = kt * 16 + (lane & 15);
      for (int i = 0; i < 8; ++i) {
        int row  = rt * 16 + ((lane >> 4) ? i + 8 : i);
        int qidx = q0 + row;
        float val = s[i] * SCALE;
        if (kcol > qidx) val = -INFINITY;   // causal (strict upper triangle)
        sc[row * SS + kcol] = val;
      }
    }
  }
  __syncthreads();

  // ---- row softmax stats: 4 lanes per row, shfl_xor reduce
  {
    int row = tid >> 2;
    int sub = tid & 3;
    const float* srow = sc + row * SS;
    float m = -INFINITY;
    for (int kk = sub; kk < keyLen; kk += 4) m = fmaxf(m, srow[kk]);
    m = fmaxf(m, __shfl_xor(m, 1));
    m = fmaxf(m, __shfl_xor(m, 2));
    float l = 0.0f;
    for (int kk = sub; kk < keyLen; kk += 4) l += __expf(srow[kk] - m);
    l += __shfl_xor(l, 1);
    l += __shfl_xor(l, 2);
    if (sub == 0) { sm_m[row] = m; sm_il[row] = 1.0f / l; }
  }
  __syncthreads();

  // ---- write normalized attention map for the last head
  if (hh == HH - 1) {
    float* mp = map_out + (long)b * SS * SS + (long)q0 * SS;
    for (int tt = tid; tt < 32 * SS; tt += 128) {
      int r = tt >> 10, kk = tt & (SS - 1);
      float val = 0.0f;
      if (kk < keyLen) val = __expf(sc[r * SS + kk] - sm_m[r]) * sm_il[r];
      mp[(long)r * SS + kk] = val;
    }
  }

  // ---- PV: wave -> (row half, d half); K loop over keys in steps of 32.
  //      V tile staged once per WG via CDNA5 async global->LDS copies.
  int rhalf = wave & 1, dhalf = wave >> 1;
  v8f acc0 = (v8f){}, acc1 = (v8f){};
  int arow = rhalf * 16 + (lane & 15);
  int akb  = (lane >> 4) * 8;
  float am = sm_m[arow], ail = sm_il[arow];
  const float* srow = sc + arow * SS;
  for (int kt32 = 0; kt32 < keyLen; kt32 += 32) {
    // async-stage V[kt32..kt32+63][0..31] -> Vt at row stride 64 (16B per lane)
    for (int tt = tid; tt < 512; tt += 128) {
      int r = tt >> 3, c4 = (tt & 7) << 2;
      async_copy_b128(v + (long)(b * SS + kt32 + r) * EE + hh * DD + c4,
                      &Vt[r * 64 + c4]);
    }
    async_wait0();
    __syncthreads();

    Frag ap;
    float4 f0 = *(const float4*)(srow + kt32 + akb);
    float4 f1 = *(const float4*)(srow + kt32 + akb + 4);
    float4 g0 = *(const float4*)(srow + kt32 + 16 + akb);
    float4 g1 = *(const float4*)(srow + kt32 + 16 + akb + 4);
    ap.v[0]  = (__bf16)(__expf(f0.x - am) * ail);
    ap.v[1]  = (__bf16)(__expf(f0.y - am) * ail);
    ap.v[2]  = (__bf16)(__expf(f0.z - am) * ail);
    ap.v[3]  = (__bf16)(__expf(f0.w - am) * ail);
    ap.v[4]  = (__bf16)(__expf(f1.x - am) * ail);
    ap.v[5]  = (__bf16)(__expf(f1.y - am) * ail);
    ap.v[6]  = (__bf16)(__expf(f1.z - am) * ail);
    ap.v[7]  = (__bf16)(__expf(f1.w - am) * ail);
    ap.v[8]  = (__bf16)(__expf(g0.x - am) * ail);
    ap.v[9]  = (__bf16)(__expf(g0.y - am) * ail);
    ap.v[10] = (__bf16)(__expf(g0.z - am) * ail);
    ap.v[11] = (__bf16)(__expf(g0.w - am) * ail);
    ap.v[12] = (__bf16)(__expf(g1.x - am) * ail);
    ap.v[13] = (__bf16)(__expf(g1.y - am) * ail);
    ap.v[14] = (__bf16)(__expf(g1.z - am) * ail);
    ap.v[15] = (__bf16)(__expf(g1.w - am) * ail);
    int kb8 = (lane >> 4) * 8;
    for (int nt = 0; nt < 2; ++nt) {
      Frag bvf;
      int d = dhalf * 32 + nt * 16 + (lane & 15);
      const float* vp = Vt + (long)kb8 * 64 + d;
      for (int i = 0; i < 16; ++i) bvf.v[i] = (__bf16)vp[(i < 8 ? i : i + 8) * 64];
      if (nt == 0) acc0 = wmma_bf16(ap, bvf, acc0);
      else         acc1 = wmma_bf16(ap, bvf, acc1);
    }
    __syncthreads();   // before next iteration overwrites Vt
  }
  // residual add of context into h (each element uniquely owned by this WG)
  for (int nt = 0; nt < 2; ++nt) {
    int col = dhalf * 32 + nt * 16 + (lane & 15);
    for (int i = 0; i < 8; ++i) {
      int row  = rhalf * 16 + ((lane >> 4) ? i + 8 : i);
      int qidx = q0 + row;
      float cval = (nt == 0) ? acc0[i] : acc1[i];
      float* hp = hres + ((long)(b * SS + qidx)) * EE + hh * DD + col;
      *hp += cval;
    }
  }
}

// ---------------- residual add ----------------
__global__ __launch_bounds__(256) void add_kernel(float* __restrict__ a,
                                                  const float* __restrict__ b) {
  long n = (long)blockIdx.x * 256 + threadIdx.x;
  a[n] += b[n];
}

// ---------------- sum-pool over S ----------------
__global__ __launch_bounds__(256) void pool_kernel(const float* __restrict__ h,
                                                   float* __restrict__ pooled) {
  int n = blockIdx.x * 256 + threadIdx.x;  // over B*E
  int b = n / EE, e = n % EE;
  float s = 0.0f;
  const float* p = h + (long)b * SS * EE + e;
  for (int i = 0; i < SS; ++i) s += p[(long)i * EE];
  pooled[n] = s;
}

// ---------------- host orchestration ----------------
extern "C" void kernel_launch(void* const* d_in, const int* in_sizes, int n_in,
                              void* d_out, int out_size, void* d_ws, size_t ws_size,
                              hipStream_t stream) {
  const int*   x    = (const int*)d_in[0];
  const float* emb  = (const float*)d_in[1];
  const float* ln_g = (const float*)d_in[2];
  const float* ln_b = (const float*)d_in[3];
  const float* Wq   = (const float*)d_in[4];
  const float* bq   = (const float*)d_in[5];
  const float* Wk   = (const float*)d_in[6];
  const float* bk   = (const float*)d_in[7];
  const float* Wv   = (const float*)d_in[8];
  const float* bv   = (const float*)d_in[9];
  const float* Wff  = (const float*)d_in[10];
  const float* bff  = (const float*)d_in[11];
  const float* W1   = (const float*)d_in[12];
  const float* b1   = (const float*)d_in[13];
  const float* W2   = (const float*)d_in[14];
  const float* b2   = (const float*)d_in[15];
  float* out = (float*)d_out;

  const long BSE = (long)BB * SS * EE;
  float* h      = (float*)d_ws;
  float* xn     = h + BSE;
  float* qb     = xn + BSE;
  float* kb     = qb + BSE;
  float* vb     = kb + BSE;
  float* pooled = vb + BSE;          // [B,E]
  float* hidden = pooled + BB * EE;  // [B,NH]

  embed_kernel<<<(int)(BSE / 256), 256, 0, stream>>>(x, emb, h);

  const dim3 gProj(EE / 64, (BB * SS) / 64);    // 8 x 128
  const size_t attnShm = (size_t)(32 * SS + 64 + 32 * 64) * sizeof(float);  // ~136 KB
  for (int l = 0; l < LL; ++l) {
    ln_kernel<<<BB * SS, 256, 0, stream>>>(h, ln_g + (long)l * EE, ln_b + (long)l * EE, xn);

    gemm_kernel<1, true, false><<<gProj, 128, 0, stream>>>(
        xn, Wq + (long)l * HH * EE * DD, bq + (long)l * HH * DD, qb, BB * SS, EE, EE);
    gemm_kernel<1, true, false><<<gProj, 128, 0, stream>>>(
        xn, Wk + (long)l * HH * EE * DD, bk + (long)l * HH * DD, kb, BB * SS, EE, EE);
    gemm_kernel<1, true, false><<<gProj, 128, 0, stream>>>(
        xn, Wv + (long)l * HH * EE * DD, bv + (long)l * HH * DD, vb, BB * SS, EE, EE);

    float* map_out = out + 256000 + (long)l * BB * SS * SS;
    attn_kernel<<<dim3(SS / 32, HH, BB), 128, attnShm, stream>>>(qb, kb, vb, h, map_out);

    gemm_kernel<0, true, false><<<gProj, 128, 0, stream>>>(
        h, Wff + (long)l * EE * EE, bff + (long)l * EE, qb, BB * SS, EE, EE);
    add_kernel<<<(int)(BSE / 256), 256, 0, stream>>>(h, qb);
  }

  pool_kernel<<<(BB * EE) / 256, 256, 0, stream>>>(h, pooled);
  gemm_kernel<0, true, true><<<dim3(NHH / 64, 1), 128, 0, stream>>>(
      pooled, W1, b1, hidden, BB, NHH, EE);
  gemm_kernel<0, true, false><<<dim3(VV / 64, 1), 128, 0, stream>>>(
      hidden, W2, b2, out, BB, VV, NHH);
}
